// CrissCrossAttention3D2_6047313953132
// MI455X (gfx1250) — matrix operand order repaired and checked by reading, not compiled, not run on hardware
//
#include <hip/hip_runtime.h>
#include <hip/hip_bf16.h>

// CDNA5 / gfx1250 criss-cross 3D attention.
// Strategy: all contractions on v_wmma_f32_16x16x32_f16 (K=32 == Cq exactly),
// fp32 accumulate, fp32 softmax. Logits staged in d_ws (127.4 MB, fits in the
// 192MB L2 so the softmax pass is L2-resident).

typedef __attribute__((ext_vector_type(16))) _Float16 v16h;
typedef __attribute__((ext_vector_type(8)))  float    v8f;

#define S   48
#define S2  (48 * 48)
#define S3  (48 * 48 * 48)
#define CQ  32
#define CV  64
#define NB  2
#define XT  (3 * S)          // 144 logits per voxel

// ---------------------------------------------------------------------------
// Logits: one wave per line-GEMM  E[m,n] = sum_c Q[c, vox(m)] * K[c, vox(n)]
// M=N=48 (positions along the axis), K=32 (channels). 9 WMMAs per wave.
// Axis is encoded via strides: fixed coords (i,j) with strides (si,sj),
// moving coord stride m_stride. Writes E into L[vox*144 + axis_off + n].
// ---------------------------------------------------------------------------
__global__ __launch_bounds__(32) void cca3d_logits_wmma(
    const float* __restrict__ Q, const float* __restrict__ Kt,
    float* __restrict__ L,
    int si, int sj, int m_stride, int axis_off, int mask_diag)
{
  const int line = blockIdx.x;                 // [0, NB*S*S)
  const int b    = line / (S * S);
  const int rr   = line - b * (S * S);
  const int fi   = rr / S;
  const int fj   = rr - fi * S;
  const int base = fi * si + fj * sj;          // voxel-local offset of the line

  const float* __restrict__ Qb = Q  + (size_t)b * CQ * S3 + base;
  const float* __restrict__ Kb = Kt + (size_t)b * CQ * S3 + base;

  const int lane = threadIdx.x & 31;
  const int l16  = lane & 15;
  const int hi   = lane >> 4;                  // 0: lanes 0-15, 1: lanes 16-31

  // A fragments (16x32 f16): lane row m = mt*16 + l16.
  // Element e -> k: lanes 0-15 hold K {0..7,16..23}, lanes 16-31 hold {8..15,24..31}.
  v16h afrag[3];
  #pragma unroll
  for (int mt = 0; mt < 3; ++mt) {
    const int m = mt * 16 + l16;
    const float* qm = Qb + (size_t)m * m_stride;
    #pragma unroll
    for (int e = 0; e < 16; ++e) {
      const int k = (hi ? 8 : 0) + (e < 8 ? e : e + 8);
      afrag[mt][e] = (_Float16)qm[(size_t)k * S3];
    }
  }
  // B fragments (32x16 f16): column n = nt*16 + l16; lanes 0-15 hold K {0..7,16..23},
  // lanes 16-31 hold K {8..15,24..31} (same order as A).
  v16h bfrag[3];
  #pragma unroll
  for (int nt = 0; nt < 3; ++nt) {
    const int n = nt * 16 + l16;
    const float* kn = Kb + (size_t)n * m_stride;
    #pragma unroll
    for (int e = 0; e < 16; ++e) {
      const int k = (hi ? 8 : 0) + (e < 8 ? e : e + 8);
      bfrag[nt][e] = (_Float16)kn[(size_t)k * S3];
    }
  }

  v8f acc[3][3];
  #pragma unroll
  for (int mt = 0; mt < 3; ++mt)
    #pragma unroll
    for (int nt = 0; nt < 3; ++nt)
      acc[mt][nt] = (v8f){0.f, 0.f, 0.f, 0.f, 0.f, 0.f, 0.f, 0.f};

  #pragma unroll
  for (int mt = 0; mt < 3; ++mt)
    #pragma unroll
    for (int nt = 0; nt < 3; ++nt)
      acc[mt][nt] = __builtin_amdgcn_wmma_f32_16x16x32_f16(
          false, afrag[mt], false, bfrag[nt], (short)0, acc[mt][nt],
          false, false);

  // C/D layout: VGPR r -> row m = mt*16 + r + hi*8, col n = nt*16 + l16.
  const size_t vox_base = (size_t)b * S3 + base;
  const float neg_inf = -__builtin_inff();
  #pragma unroll
  for (int mt = 0; mt < 3; ++mt)
    #pragma unroll
    for (int nt = 0; nt < 3; ++nt)
      #pragma unroll
      for (int r = 0; r < 8; ++r) {
        const int m = mt * 16 + r + hi * 8;
        const int n = nt * 16 + l16;
        float v = acc[mt][nt][r];
        if (mask_diag && (m == n)) v = neg_inf;   // -inf diagonal (W/D axes)
        L[(vox_base + (size_t)m * m_stride) * XT + axis_off + n] = v;
      }
}

// ---------------------------------------------------------------------------
// Softmax over 144 contiguous logits per voxel. One wave32 per voxel.
// ---------------------------------------------------------------------------
__global__ __launch_bounds__(256) void cca3d_softmax(float* __restrict__ L)
{
  const int wave = blockIdx.x * (blockDim.x >> 5) + (threadIdx.x >> 5);
  const int lane = threadIdx.x & 31;
  float* __restrict__ row = L + (size_t)wave * XT;

  float vals[5];
  float mx = -__builtin_inff();
  #pragma unroll
  for (int t = 0; t < 5; ++t) {
    const int idx = lane + 32 * t;
    vals[t] = (idx < XT) ? row[idx] : -__builtin_inff();
    mx = fmaxf(mx, vals[t]);
  }
  #pragma unroll
  for (int off = 16; off > 0; off >>= 1)
    mx = fmaxf(mx, __shfl_xor(mx, off, 32));

  float sum = 0.f;
  #pragma unroll
  for (int t = 0; t < 5; ++t) {
    const int idx = lane + 32 * t;
    vals[t] = (idx < XT) ? __expf(vals[t] - mx) : 0.f;   // exp(-inf)=0 masks
    sum += vals[t];
  }
  #pragma unroll
  for (int off = 16; off > 0; off >>= 1)
    sum += __shfl_xor(sum, off, 32);

  const float inv = 1.f / sum;
  #pragma unroll
  for (int t = 0; t < 5; ++t) {
    const int idx = lane + 32 * t;
    if (idx < XT) row[idx] = vals[t] * inv;
  }
}

// ---------------------------------------------------------------------------
// Output: one wave per line.  O[m, c] (+)= sum_x att[m, x] * V[c, vox(x)]
// M=48 positions, N=64 channels, K=48 split as 32 + 16(zero-padded). 24 WMMAs.
// Pass H stores; passes W/D accumulate (exclusive ownership within a pass,
// passes serialized on the stream).
// ---------------------------------------------------------------------------
__global__ __launch_bounds__(32) void cca3d_out_wmma(
    const float* __restrict__ L, const float* __restrict__ V,
    float* __restrict__ out,
    int si, int sj, int m_stride, int axis_off, int accumulate)
{
  const int line = blockIdx.x;
  const int b    = line / (S * S);
  const int rr   = line - b * (S * S);
  const int fi   = rr / S;
  const int fj   = rr - fi * S;
  const int base = fi * si + fj * sj;

  const size_t vox_base = (size_t)b * S3 + base;
  const float* __restrict__ Vb = V + (size_t)b * CV * S3 + base;
  float* __restrict__ Ob = out + (size_t)b * CV * S3 + base;

  const int lane = threadIdx.x & 31;
  const int l16  = lane & 15;
  const int hi   = lane >> 4;

  v8f acc[3][4];
  #pragma unroll
  for (int mt = 0; mt < 3; ++mt)
    #pragma unroll
    for (int nt = 0; nt < 4; ++nt)
      acc[mt][nt] = (v8f){0.f, 0.f, 0.f, 0.f, 0.f, 0.f, 0.f, 0.f};

  #pragma unroll
  for (int kt = 0; kt < 2; ++kt) {
    // A = attention probs (f16), A[m, kx] = L[vox(m)*144 + axis_off + kx]
    v16h afrag[3];
    #pragma unroll
    for (int mt = 0; mt < 3; ++mt) {
      const int m = mt * 16 + l16;
      const float* lrow = L + (vox_base + (size_t)m * m_stride) * XT + axis_off;
      #pragma unroll
      for (int e = 0; e < 16; ++e) {
        const int kl = (hi ? 8 : 0) + (e < 8 ? e : e + 8);
        const int kx = kt * 32 + kl;
        afrag[mt][e] = (kx < S) ? (_Float16)lrow[kx] : (_Float16)0.f;
      }
    }
    // B = V (f16), B[kx, n=c] = V[b, c, vox(kx)]
    v16h bfrag[4];
    #pragma unroll
    for (int nt = 0; nt < 4; ++nt) {
      const int c = nt * 16 + l16;
      const float* vc = Vb + (size_t)c * S3;
      #pragma unroll
      for (int e = 0; e < 16; ++e) {
        const int kl = (hi ? 8 : 0) + (e < 8 ? e : e + 8);
        const int kx = kt * 32 + kl;
        bfrag[nt][e] = (kx < S) ? (_Float16)vc[(size_t)kx * m_stride]
                                : (_Float16)0.f;
      }
    }
    #pragma unroll
    for (int mt = 0; mt < 3; ++mt)
      #pragma unroll
      for (int nt = 0; nt < 4; ++nt)
        acc[mt][nt] = __builtin_amdgcn_wmma_f32_16x16x32_f16(
            false, afrag[mt], false, bfrag[nt], (short)0, acc[mt][nt],
            false, false);
  }

  #pragma unroll
  for (int mt = 0; mt < 3; ++mt)
    #pragma unroll
    for (int nt = 0; nt < 4; ++nt)
      #pragma unroll
      for (int r = 0; r < 8; ++r) {
        const int m = mt * 16 + r + hi * 8;
        const int c = nt * 16 + l16;
        const size_t idx = (size_t)c * S3 + (size_t)m * m_stride;
        const float v = acc[mt][nt][r];
        Ob[idx] = accumulate ? (Ob[idx] + v) : v;
      }
}

// ---------------------------------------------------------------------------
extern "C" void kernel_launch(void* const* d_in, const int* in_sizes, int n_in,
                              void* d_out, int out_size, void* d_ws, size_t ws_size,
                              hipStream_t stream) {
  const float* Q = (const float*)d_in[0];   // (2,32,48,48,48)
  const float* K = (const float*)d_in[1];   // (2,32,48,48,48)
  const float* V = (const float*)d_in[2];   // (2,64,48,48,48)
  float* out = (float*)d_out;               // (2,64,48,48,48)
  float* L   = (float*)d_ws;                // needs 2*48^3*144*4 = 127.4 MB

  const dim3 lines(NB * S * S);             // 4608 line-GEMMs per axis

  // Logits.  axis H: fixed (w,d), move h (stride S2), no diag mask.
  cca3d_logits_wmma<<<lines, 32, 0, stream>>>(Q, K, L, S,  1, S2, 0,     0);
  //         axis W: fixed (h,d), move w (stride S), -inf diag.
  cca3d_logits_wmma<<<lines, 32, 0, stream>>>(Q, K, L, S2, 1, S,  S,     1);
  //         axis D: fixed (h,w), move d (stride 1), -inf diag.
  cca3d_logits_wmma<<<lines, 32, 0, stream>>>(Q, K, L, S2, S, 1,  2 * S, 1);

  // Softmax over 144 per voxel: 8 waves / 256-thread block, one wave / voxel.
  const int nvox = NB * S3;                 // 221184, divisible by 8
  cca3d_softmax<<<nvox / 8, 256, 0, stream>>>(L);

  // Outputs: H writes, W and D accumulate.
  cca3d_out_wmma<<<lines, 32, 0, stream>>>(L, V, out, S,  1, S2, 0,     0);
  cca3d_out_wmma<<<lines, 32, 0, stream>>>(L, V, out, S2, 1, S,  S,     1);
  cca3d_out_wmma<<<lines, 32, 0, stream>>>(L, V, out, S2, S, 1,  2 * S, 1);
}
